// BaseVectorQuantizer_29334626631742
// MI455X (gfx1250) — compile-verified, matrix operand order checked by reading, not run on hardware
//
#include <hip/hip_runtime.h>

typedef __attribute__((ext_vector_type(16))) _Float16 v16h;
typedef __attribute__((ext_vector_type(8)))  float    v8f;
typedef __attribute__((ext_vector_type(4)))  float    v4f;

// ---------------------------------------------------------------------------
// Kernel A: precompute table[k] = LN(relu(codebook[k] @ W1 + b1) @ W2 + b2)
// for all K=1024 codebook rows. One block = 16 rows. The 256-deep GEMM runs
// on v_wmma_f32_16x16x32_f16 (f16 in, f32 accumulate).
// ---------------------------------------------------------------------------
__global__ __launch_bounds__(256) void vq_mlp_table(
    const float* __restrict__ codebook,   // [1024, 8]
    const float* __restrict__ W1,         // [8, 256]
    const float* __restrict__ b1,         // [256]
    const float* __restrict__ W2,         // [256, 256]
    const float* __restrict__ b2,         // [256]
    const float* __restrict__ gamma,      // [256]
    const float* __restrict__ beta,       // [256]
    float* __restrict__ table)            // [1024, 256]
{
    __shared__ _Float16 hH[16 * 256];     // h = relu(cb@W1+b1), f16, 8 KB
    __shared__ float    q[16 * 256];      // q = h@W2+b2, f32, 16 KB
    __shared__ float    mu_s[16], rs_s[16];

    const int t       = threadIdx.x;
    const int rowBase = blockIdx.x * 16;

    // ---- phase 0: h[16][256] = relu(codebook_tile @ W1 + b1), K=8 on VALU ----
    #pragma unroll
    for (int j = 0; j < 16; ++j) {
        const int idx = t * 16 + j;
        const int r = idx >> 8, c = idx & 255;
        float acc = b1[c];
        #pragma unroll
        for (int d = 0; d < 8; ++d)
            acc = fmaf(codebook[(rowBase + r) * 8 + d], W1[d * 256 + c], acc);
        hH[idx] = (_Float16)fmaxf(acc, 0.0f);
    }
    __syncthreads();

    // ---- phase 1: q = h @ W2 via WMMA. 8 waves x 2 column-tiles = 256 cols ----
    const int lane  = t & 31;
    const int wave  = t >> 5;
    const int mrow  = lane & 15;          // A-matrix row owned by this lane
    const int kHalf = (lane >> 4) << 3;   // lanes 16-31 hold K+8 block (ISA 7.12.2)

    #pragma unroll
    for (int tile = 0; tile < 2; ++tile) {
        const int colBase = (wave + tile * 8) * 16;
        const int col     = colBase + (lane & 15);
        v8f acc = {};
        #pragma unroll
        for (int kb = 0; kb < 256; kb += 32) {
            v16h a, b;
            #pragma unroll
            for (int v = 0; v < 8; ++v) {
                // 16-bit A layout: VGPR 0-3 -> K = kHalf+2v..; VGPR 4-7 -> K+16
                const int kOff = kb + kHalf + ((v < 4) ? (2 * v) : (16 + 2 * (v - 4)));
                a[2 * v]     = hH[mrow * 256 + kOff];
                a[2 * v + 1] = hH[mrow * 256 + kOff + 1];
                // B (32x16): lane owns column `col`, same K packing
                b[2 * v]     = (_Float16)W2[(size_t)kOff * 256 + col];
                b[2 * v + 1] = (_Float16)W2[(size_t)(kOff + 1) * 256 + col];
            }
            acc = __builtin_amdgcn_wmma_f32_16x16x32_f16(
                false, a, false, b, (short)0, acc, false, false);
        }
        // C/D layout: VGPR v holds M = v + 8*(lane>>4), N = lane&15
        #pragma unroll
        for (int v = 0; v < 8; ++v) {
            const int m = v + ((lane >> 4) << 3);
            q[m * 256 + col] = acc[v] + b2[col];
        }
    }
    __syncthreads();

    // ---- phase 2: LayerNorm(16 rows of 256) ----
    if (t < 16) {
        float s = 0.0f, s2 = 0.0f;
        for (int c = 0; c < 256; ++c) {
            const float x = q[t * 256 + c];
            s += x; s2 += x * x;
        }
        const float mu  = s * (1.0f / 256.0f);
        const float var = s2 * (1.0f / 256.0f) - mu * mu;
        mu_s[t] = mu;
        rs_s[t] = rsqrtf(var + 1e-5f);
    }
    __syncthreads();

    #pragma unroll
    for (int j = 0; j < 16; ++j) {
        const int idx = t * 16 + j;
        const int r = idx >> 8, c = idx & 255;
        table[(size_t)(rowBase + r) * 256 + c] =
            (q[idx] - mu_s[r]) * rs_s[r] * gamma[c] + beta[c];
    }
}

// ---------------------------------------------------------------------------
// Kernel B: per 256-row tile: argmin over 1024 codes (codebook in LDS,
// broadcast reads), then streaming nontemporal writes of the one-hot rows
// (256 MB total) and the gathered precomputed table rows (64 MB total).
// ---------------------------------------------------------------------------
__global__ __launch_bounds__(256) void vq_assign_write(
    const float* __restrict__ features,   // [N, 8]
    const float* __restrict__ codebook,   // [1024, 8]
    const float* __restrict__ table,      // [1024, 256] (hot in L2)
    float* __restrict__ outQ,             // [N, 256]
    int*   __restrict__ outIdx,           // [N]
    float* __restrict__ outOneHot)        // [N, 1024]
{
    __shared__ float cb[1024 * 8];        // 32 KB
    __shared__ int   idx_s[256];

    const int t       = threadIdx.x;
    const int rowBase = blockIdx.x * 256;

    // stage codebook into LDS (2048 x b128)
    {
        const v4f* src = (const v4f*)codebook;
        v4f*       dst = (v4f*)cb;
        #pragma unroll
        for (int i = 0; i < 8; ++i)
            dst[t + 256 * i] = src[t + 256 * i];
    }

    // this thread's feature row
    float f[8];
    {
        const v4f* fr = (const v4f*)(features + (size_t)(rowBase + t) * 8);
        const v4f f0 = fr[0], f1 = fr[1];
        f[0] = f0.x; f[1] = f0.y; f[2] = f0.z; f[3] = f0.w;
        f[4] = f1.x; f[5] = f1.y; f[6] = f1.z; f[7] = f1.w;
    }
    __syncthreads();

    // argmin_k ||f - cb[k]||^2 ; strict < keeps first minimum (jnp.argmin)
    float best = 3.402823466e38f;
    int   bi   = 0;
    for (int k = 0; k < 1024; ++k) {
        const float* c = &cb[k * 8];
        float d  = f[0] - c[0];
        float s  = d * d;
        #pragma unroll
        for (int j = 1; j < 8; ++j) {
            d = f[j] - c[j];
            s = fmaf(d, d, s);
        }
        if (s < best) { best = s; bi = k; }
    }
    idx_s[t] = bi;
    outIdx[rowBase + t] = bi;
    __syncthreads();

    // one-hot rows: 4 KB per row, coalesced b128 nontemporal streaming stores
    for (int r = 0; r < 256; ++r) {
        const int code = idx_s[r];
        v4f w = {0.0f, 0.0f, 0.0f, 0.0f};
        if ((code >> 2) == t) w[code & 3] = 1.0f;
        v4f* dst = (v4f*)(outOneHot + (size_t)(rowBase + r) * 1024) + t;
        __builtin_nontemporal_store(w, dst);
    }

    // gather final LN'ed rows from the table: 4 rows / iteration, b128
    const int rSub  = t >> 6;     // 0..3
    const int cQuad = t & 63;     // 0..63
    for (int rb = 0; rb < 256; rb += 4) {
        const int r = rb + rSub;
        const v4f* src = (const v4f*)(table + (size_t)idx_s[r] * 256) + cQuad;
        const v4f v = *src;
        v4f* dst = (v4f*)(outQ + (size_t)(rowBase + r) * 256) + cQuad;
        __builtin_nontemporal_store(v, dst);
    }
}

// ---------------------------------------------------------------------------
extern "C" void kernel_launch(void* const* d_in, const int* in_sizes, int n_in,
                              void* d_out, int out_size, void* d_ws, size_t ws_size,
                              hipStream_t stream) {
    const float* features = (const float*)d_in[0];
    const float* codebook = (const float*)d_in[1];
    const float* W1       = (const float*)d_in[2];
    const float* b1       = (const float*)d_in[3];
    const float* W2       = (const float*)d_in[4];
    const float* b2       = (const float*)d_in[5];
    const float* gamma    = (const float*)d_in[6];
    const float* beta     = (const float*)d_in[7];

    const int N = in_sizes[0] / 8;   // 65536 tokens
    const int K = in_sizes[1] / 8;   // 1024 codes

    float* table = (float*)d_ws;     // [K, 256] = 1 MB scratch

    // d_out layout: quantized_out [N,256] f32 | indices [N] i32 | one-hot [N,K] f32
    float* outQ   = (float*)d_out;
    int*   outIdx = (int*)((float*)d_out + (size_t)N * 256);
    float* outOH  = (float*)d_out + (size_t)N * 256 + (size_t)N;

    vq_mlp_table<<<K / 16, 256, 0, stream>>>(codebook, W1, b1, W2, b2,
                                             gamma, beta, table);
    vq_assign_write<<<N / 256, 256, 0, stream>>>(features, codebook, table,
                                                 outQ, outIdx, outOH);
}
